// RelationalMultiHeadAttentionMP_45157286150354
// MI455X (gfx1250) — matrix operand order repaired and checked by reading, not compiled, not run on hardware
//
#include <hip/hip_runtime.h>

// ---------------- problem constants ----------------
#define N_NODES   100000
#define T_TYPES   4
#define E_EDGES   150000
#define HID       128
#define NH        8
#define DH        16
#define MS        128            // NH*DH
#define TILES_PER_TYPE (E_EDGES / 16)   // 9375 (E divisible by 16)
#define TOTAL_EDGES (T_TYPES * E_EDGES) // 600000
#define RSTRIDE   68             // padded red-buffer row stride (floats) to dodge bank conflicts

typedef __attribute__((ext_vector_type(16))) __bf16 v16bf;
typedef __attribute__((ext_vector_type(8)))  float  v8f;

struct __align__(16) U4 { unsigned x, y, z, w; };
struct __align__(16) F4 { float x, y, z, w; };
union AFrag { v16bf v; U4 q[2]; };
union BFrag { v16bf v; U4 q[2]; };

// ---------------- helpers ----------------
__device__ __forceinline__ unsigned short f2bf(float f) {
    union { float f; unsigned u; } v; v.f = f;
    unsigned u = v.u;
    u += 0x7fffu + ((u >> 16) & 1u);   // round-to-nearest-even
    return (unsigned short)(u >> 16);
}

// monotone float -> orderable uint encoding (for atomicMax-based segment max)
__device__ __forceinline__ unsigned fenc(float f) {
    union { float f; unsigned u; } v; v.f = f;
    return (v.u & 0x80000000u) ? ~v.u : (v.u | 0x80000000u);
}
__device__ __forceinline__ float fdec(unsigned k) {
    union { float f; unsigned u; } v;
    v.u = (k & 0x80000000u) ? (k ^ 0x80000000u) : ~k;
    return v.f;
}

// ---------------- prep kernels ----------------
__global__ void init_kernel(float* __restrict__ out, float* __restrict__ denom,
                            unsigned* __restrict__ nmax) {
    size_t i = (size_t)blockIdx.x * blockDim.x + threadIdx.x;
    size_t stride = (size_t)gridDim.x * blockDim.x;
    for (size_t j = i; j < (size_t)N_NODES * MS; j += stride) out[j] = 0.f;
    for (size_t j = i; j < (size_t)N_NODES * NH; j += stride) { denom[j] = 0.f; nmax[j] = 0u; }
}

__global__ void cvt_x_kernel(const float* __restrict__ x, unsigned short* __restrict__ xbf) {
    size_t i = (size_t)blockIdx.x * blockDim.x + threadIdx.x;
    if (i < (size_t)N_NODES * HID) xbf[i] = f2bf(x[i]);
}

__global__ void cvt_w_kernel(const float* __restrict__ Wq, const float* __restrict__ Wk,
                             const float* __restrict__ Wm,
                             unsigned short* __restrict__ wq, unsigned short* __restrict__ wk,
                             unsigned short* __restrict__ wm) {
    int i = blockIdx.x * blockDim.x + threadIdx.x;
    const int nq = T_TYPES * HID * MS;        // 65536
    const int nm = T_TYPES * 2 * HID * MS;    // 131072
    if (i < nq)               wq[i] = f2bf(Wq[i]);
    else if (i < 2 * nq)      wk[i - nq] = f2bf(Wk[i - nq]);
    else if (i < 2 * nq + nm) wm[i - 2 * nq] = f2bf(Wm[i - 2 * nq]);
}

// ---------------- phase 1: per-edge attention scores via WMMA ----------------
__global__ __launch_bounds__(32) void score_kernel(
    const unsigned short* __restrict__ xbf,   // [N,128] bf16
    const int* __restrict__ adj,              // [T,E,2]
    const unsigned short* __restrict__ Wqbf,  // [T,128,128] bf16, row-major [K][N]
    const unsigned short* __restrict__ Wkbf,  // [T,128,128]
    float* __restrict__ scores)               // [T*E, 8]
{
    const int blk  = blockIdx.x;
    const int t    = blk / TILES_PER_TYPE;
    const int tile = blk % TILES_PER_TYPE;
    const int ebase = t * E_EDGES + tile * 16;   // flat edge index base
    const int lane = threadIdx.x;

    // 8704B scratch: first used as the two bf16 tiles (8KB), then reused as the
    // f32 reduction buffer (A-fragments live in registers by then).
    __shared__ __attribute__((aligned(16))) unsigned char smem[32 * RSTRIDE * 4];
    unsigned short* ldsS = (unsigned short*)smem;            // [16][128] bf16
    unsigned short* ldsT = (unsigned short*)(smem + 4096);   // [16][128] bf16
    float* red = (float*)smem;                               // [32][RSTRIDE] f32
    __shared__ int idxs[16], idxt[16];

    const unsigned short* wq = Wqbf + (size_t)t * HID * MS;
    const unsigned short* wk = Wkbf + (size_t)t * HID * MS;
    __builtin_prefetch((const char*)wq + lane * 256, 0, 3);  // warm per-type weights
    __builtin_prefetch((const char*)wk + lane * 256, 0, 3);

    if (lane < 16) {
        idxs[lane] = adj[(size_t)(ebase + lane) * 2 + 0];
        idxt[lane] = adj[(size_t)(ebase + lane) * 2 + 1];
    }
    __syncthreads();

    {   // gather 16 rows x 128 bf16 (each lane: half a row = 128B as 8x16B)
        int r = lane & 15, half = lane >> 4;
        const U4* gS = (const U4*)(xbf + (size_t)idxs[r] * HID + half * 64);
        const U4* gT = (const U4*)(xbf + (size_t)idxt[r] * HID + half * 64);
        U4* dS = (U4*)(ldsS + r * HID + half * 64);
        U4* dT = (U4*)(ldsT + r * HID + half * 64);
#pragma unroll
        for (int i = 0; i < 8; ++i) { dS[i] = gS[i]; dT[i] = gT[i]; }
    }
    __syncthreads();

    const int m = lane & 15;         // A-fragment row
    const int khalf = lane >> 4;     // A-fragment K-half

    // pull all A-fragments into registers so LDS can be reused
    AFrag aT[4], aS[4];
#pragma unroll
    for (int ks = 0; ks < 4; ++ks) {
        const int K0 = ks * 32;
        aT[ks].q[0] = *(const U4*)(ldsT + m * HID + K0 + khalf * 8);
        aT[ks].q[1] = *(const U4*)(ldsT + m * HID + K0 + 16 + khalf * 8);
        aS[ks].q[0] = *(const U4*)(ldsS + m * HID + K0 + khalf * 8);
        aS[ks].q[1] = *(const U4*)(ldsS + m * HID + K0 + 16 + khalf * 8);
    }
    __syncthreads();   // tiles consumed; smem becomes the reduction buffer

    v8f prodH[8];      // per-head elementwise q*k fragments
#pragma unroll
    for (int head = 0; head < NH; ++head) {
        const int n0 = head * DH;
        // batch all B loads for this head (single clause; overlaps with prev WMMAs)
        BFrag bq[4], bk[4];
#pragma unroll
        for (int ks = 0; ks < 4; ++ks) {
            const int K0 = ks * 32;
            const int nn = n0 + (lane & 15);
#pragma unroll
            for (int j = 0; j < 16; ++j) {
                const int kk = K0 + (j < 8 ? khalf * 8 + j : 16 + khalf * 8 + (j - 8));
                bq[ks].v[j] = __builtin_bit_cast(__bf16, wq[(size_t)kk * MS + nn]);
                bk[ks].v[j] = __builtin_bit_cast(__bf16, wk[(size_t)kk * MS + nn]);
            }
        }
        v8f qacc = {}; v8f kacc = {};
#pragma unroll
        for (int ks = 0; ks < 4; ++ks) {
            qacc = __builtin_amdgcn_wmma_f32_16x16x32_bf16(false, aT[ks].v, false, bq[ks].v,
                                                           (short)0, qacc, false, false);
            kacc = __builtin_amdgcn_wmma_f32_16x16x32_bf16(false, aS[ks].v, false, bk[ks].v,
                                                           (short)0, kacc, false, false);
        }
        prodH[head] = qacc * kacc;
    }

    // transpose-reduce via LDS: store [lane][r][head] as two float4s per r
#pragma unroll
    for (int r = 0; r < 8; ++r) {
        F4 lo = { prodH[0][r], prodH[1][r], prodH[2][r], prodH[3][r] };
        F4 hi = { prodH[4][r], prodH[5][r], prodH[6][r], prodH[7][r] };
        *(F4*)(red + lane * RSTRIDE + r * 8)     = lo;
        *(F4*)(red + lane * RSTRIDE + r * 8 + 4) = hi;
    }
    __syncthreads();

    // lane -> (edge mm, head-quad h0); sum the 16 n-lanes; coalesced b128 store
    const int mm = lane >> 1, h0 = (lane & 1) * 4;
    const int halfm = mm >> 3, rr = mm & 7;
    F4 s = {0.f, 0.f, 0.f, 0.f};
#pragma unroll
    for (int n = 0; n < 16; ++n) {
        const F4 v = *(const F4*)(red + (n + 16 * halfm) * RSTRIDE + rr * 8 + h0);
        s.x += v.x; s.y += v.y; s.z += v.z; s.w += v.w;
    }
    s.x *= 0.25f; s.y *= 0.25f; s.z *= 0.25f; s.w *= 0.25f;   // scale = D^-0.5
    *(F4*)(scores + (size_t)(ebase + mm) * NH + h0) = s;
}

// ---------------- phase 2: segment softmax ----------------
__global__ void segmax_kernel(const float* __restrict__ scores, const int* __restrict__ adj,
                              unsigned* __restrict__ nmax) {
    int i = blockIdx.x * blockDim.x + threadIdx.x;        // over TOTAL_EDGES*NH
    if (i >= TOTAL_EDGES * NH) return;
    int eg = i >> 3, h = i & 7;
    int tgt = adj[(size_t)eg * 2 + 1];
    atomicMax(&nmax[(size_t)tgt * NH + h], fenc(scores[i]));
}

__global__ void expdenom_kernel(float* __restrict__ scores, const int* __restrict__ adj,
                                const unsigned* __restrict__ nmax, float* __restrict__ denom) {
    int i = blockIdx.x * blockDim.x + threadIdx.x;
    if (i >= TOTAL_EDGES * NH) return;
    int eg = i >> 3, h = i & 7;
    int tgt = adj[(size_t)eg * 2 + 1];
    float mv = fdec(nmax[(size_t)tgt * NH + h]);
    if (!isfinite(mv)) mv = 0.f;                          // empty-segment convention
    float e = __expf(scores[i] - mv);
    scores[i] = e;                                        // overwrite in place
    atomicAdd(&denom[(size_t)tgt * NH + h], e);
}

// ---------------- phase 3: message GEMM (WMMA) + weighted scatter ----------------
__global__ __launch_bounds__(32) void aggregate_kernel(
    const unsigned short* __restrict__ xbf,   // [N,128] bf16
    const int* __restrict__ adj,              // [T,E,2]
    const unsigned short* __restrict__ Wmbf,  // [T,256,128] bf16 row-major [K][N]
    const float* __restrict__ bm,             // [T,128]
    const float* __restrict__ evals,          // [T*E,8] exp'd scores
    const float* __restrict__ denom,          // [N,8]
    float* __restrict__ out)                  // [N,128]
{
    const int blk  = blockIdx.x;
    const int t    = blk / TILES_PER_TYPE;
    const int tile = blk % TILES_PER_TYPE;
    const int ebase = t * E_EDGES + tile * 16;
    const int lane = threadIdx.x;

    __shared__ __attribute__((aligned(16))) unsigned short ldsS[16 * HID];
    __shared__ __attribute__((aligned(16))) unsigned short ldsT[16 * HID];
    __shared__ int idxs[16], idxt[16];
    __shared__ __attribute__((aligned(16))) float probs[NH * 16];   // [head][edge]

    const unsigned short* wm = Wmbf + (size_t)t * 2 * HID * MS;
    __builtin_prefetch((const char*)wm + lane * 256, 0, 3);

    if (lane < 16) {
        idxs[lane] = adj[(size_t)(ebase + lane) * 2 + 0];
        idxt[lane] = adj[(size_t)(ebase + lane) * 2 + 1];
    }
    __syncthreads();

    {   // gather tiles
        int r = lane & 15, half = lane >> 4;
        const U4* gS = (const U4*)(xbf + (size_t)idxs[r] * HID + half * 64);
        const U4* gT = (const U4*)(xbf + (size_t)idxt[r] * HID + half * 64);
        U4* dS = (U4*)(ldsS + r * HID + half * 64);
        U4* dT = (U4*)(ldsT + r * HID + half * 64);
#pragma unroll
        for (int i = 0; i < 8; ++i) { dS[i] = gS[i]; dT[i] = gT[i]; }
    }
    // softmax probabilities, laid out [head][edge] for vector reads below
    for (int i = lane; i < NH * 16; i += 32) {
        int h = i >> 4, me = i & 15;
        float ev = evals[(size_t)(ebase + me) * NH + h];
        float d  = denom[(size_t)idxt[me] * NH + h];
        probs[i] = (d > 0.f) ? ev / d : 0.f;
    }
    __syncthreads();

    const int m = lane & 15;
    const int khalf = lane >> 4;

    // A-fragments for the concatenated K=256 ([xs | xt]) into registers
    AFrag aAll[8];
#pragma unroll
    for (int ks = 0; ks < 8; ++ks) {
        const int K0 = ks * 32;
        const unsigned short* a_lds =
            (K0 < HID) ? (ldsS + m * HID + K0) : (ldsT + m * HID + (K0 - HID));
        aAll[ks].q[0] = *(const U4*)(a_lds + khalf * 8);
        aAll[ks].q[1] = *(const U4*)(a_lds + 16 + khalf * 8);
    }

    // per-lane target rows for the scatter
    int tgtm[8];
#pragma unroll
    for (int r = 0; r < 8; ++r) tgtm[r] = idxt[r + 8 * khalf];

#pragma unroll
    for (int head = 0; head < NH; ++head) {
        const int n0 = head * DH;
        // batch all 8 B loads for this head
        BFrag b[8];
#pragma unroll
        for (int ks = 0; ks < 8; ++ks) {
            const int nn = n0 + (lane & 15);
#pragma unroll
            for (int j = 0; j < 16; ++j) {
                const int kk = ks * 32 + (j < 8 ? khalf * 8 + j : 16 + khalf * 8 + (j - 8));
                b[ks].v[j] = __builtin_bit_cast(__bf16, wm[(size_t)kk * MS + nn]);
            }
        }
        v8f acc = {};
#pragma unroll
        for (int ks = 0; ks < 8; ++ks)
            acc = __builtin_amdgcn_wmma_f32_16x16x32_bf16(false, aAll[ks].v, false, b[ks].v,
                                                          (short)0, acc, false, false);

        const float bias = bm[(size_t)t * MS + n0 + (lane & 15)];
        // probs for this head's 8 edges of this lane half: two broadcast b128 reads
        const F4 p0 = *(const F4*)(probs + head * 16 + khalf * 8);
        const F4 p1 = *(const F4*)(probs + head * 16 + khalf * 8 + 4);
        const float pv[8] = { p0.x, p0.y, p0.z, p0.w, p1.x, p1.y, p1.z, p1.w };
#pragma unroll
        for (int r = 0; r < 8; ++r) {
            float msg = acc[r] + bias;
            msg = msg > 0.f ? msg : 0.f;              // relu
            atomicAdd(&out[(size_t)tgtm[r] * MS + n0 + (lane & 15)], pv[r] * msg);
        }
    }
}

// ---------------- host launcher ----------------
extern "C" void kernel_launch(void* const* d_in, const int* in_sizes, int n_in,
                              void* d_out, int out_size, void* d_ws, size_t ws_size,
                              hipStream_t stream) {
    const float* x  = (const float*)d_in[0];
    const int*   adj = (const int*)d_in[1];
    const float* Wq = (const float*)d_in[2];
    const float* Wk = (const float*)d_in[3];
    const float* Wm = (const float*)d_in[4];
    const float* bm = (const float*)d_in[5];
    float* out = (float*)d_out;

    char* ws = (char*)d_ws;
    size_t off = 0;
    auto take = [&](size_t bytes) -> void* {
        void* p = ws + off;
        off += (bytes + 255) & ~(size_t)255;
        return p;
    };
    unsigned short* xbf   = (unsigned short*)take((size_t)N_NODES * HID * 2);
    unsigned short* wqbf  = (unsigned short*)take((size_t)T_TYPES * HID * MS * 2);
    unsigned short* wkbf  = (unsigned short*)take((size_t)T_TYPES * HID * MS * 2);
    unsigned short* wmbf  = (unsigned short*)take((size_t)T_TYPES * 2 * HID * MS * 2);
    float*          scores = (float*)take((size_t)TOTAL_EDGES * NH * 4);
    unsigned*       nmax   = (unsigned*)take((size_t)N_NODES * NH * 4);
    float*          denom  = (float*)take((size_t)N_NODES * NH * 4);

    // prep
    init_kernel<<<4096, 256, 0, stream>>>(out, denom, nmax);
    {
        size_t n = (size_t)N_NODES * HID;
        cvt_x_kernel<<<(unsigned)((n + 255) / 256), 256, 0, stream>>>(x, xbf);
    }
    {
        int n = 2 * T_TYPES * HID * MS + T_TYPES * 2 * HID * MS;  // 262144
        cvt_w_kernel<<<(n + 255) / 256, 256, 0, stream>>>(Wq, Wk, Wm, wqbf, wkbf, wmbf);
    }

    const int ntiles = T_TYPES * TILES_PER_TYPE;     // 37500 tiles of 16 edges

    // phase 1: scores
    score_kernel<<<ntiles, 32, 0, stream>>>(xbf, adj, wqbf, wkbf, scores);

    // phase 2: segment softmax
    {
        int n = TOTAL_EDGES * NH;
        segmax_kernel<<<(n + 255) / 256, 256, 0, stream>>>(scores, adj, nmax);
        expdenom_kernel<<<(n + 255) / 256, 256, 0, stream>>>(scores, adj, nmax, denom);
    }

    // phase 3: messages + weighted scatter
    aggregate_kernel<<<ntiles, 32, 0, stream>>>(xbf, adj, wmbf, bm, scores, denom, out);
}
